// FujiGatedDeltaNet_71159018160279
// MI455X (gfx1250) — compile-verified, matrix operand order checked by reading, not run on hardware
//
#include <hip/hip_runtime.h>
#include <hip/hip_bf16.h>
#include <stdint.h>

// ---------------------------------------------------------------- types
typedef __bf16 bf16_t;
typedef __attribute__((ext_vector_type(16))) __bf16 v16bf;
typedef __attribute__((ext_vector_type(8)))  float  v8f;

union AFrag { v16bf v; unsigned u[8]; };
union CFrag { v8f  v; float    f[8]; };

// ---------------------------------------------------------------- sizes
#define BB      2
#define SS      4096
#define HH      2048
#define NH      16
#define DKD     128
#define DVD     128
#define KCONV   4
#define CHKSZ   64
#define NCHUNK  (SS / CHKSZ)
#define NTOK    (BB * SS)
#define QKVZN   8192
#define CONVC   6144

// ---------------------------------------------------------------- WMMA helpers
// A/B operand fragment (16x32 bf16) from LDS tile stored row-major [row][k],
// even element stride `ld`. Per ISA 7.12.2 (16-bit A layout):
// lane L holds row (L&15); K pairs per VGPR v:
//   v<4 : dword v     + 4*(L>>4)
//   v>=4: dword v + 4 + 4*(L>>4)
__device__ inline v16bf ld_frag(const bf16_t* base, int ld, int row0, int k0) {
  const int lane = threadIdx.x & 31;
  const int grp  = lane >> 4;
  const unsigned* p = (const unsigned*)(base + (row0 + (lane & 15)) * ld + k0);
  AFrag f;
#pragma unroll
  for (int v = 0; v < 8; ++v) {
    const int dw = (v < 4) ? (v + grp * 4) : (v + 4 + grp * 4);
    f.u[v] = p[dw];
  }
  return f.v;
}

__device__ inline v8f wmma_bf16(v16bf a, v16bf b, v8f c) {
  return __builtin_amdgcn_wmma_f32_16x16x32_bf16(false, a, false, b,
                                                 (short)0, c, false, false);
}

// CDNA5 async global->LDS copy (16B per lane), tracked by ASYNCcnt.
__device__ inline void async_ld_b128(unsigned lds_off, const void* gaddr) {
  asm volatile("global_load_async_to_lds_b128 %0, %1, off"
               :: "v"(lds_off), "v"(gaddr) : "memory");
}
__device__ inline unsigned lds_addr_of(const void* p) {
  return (unsigned)(uintptr_t)p;   // generic LDS address: low 32 bits = LDS offset
}

// ---------------------------------------------------------------- bf16 convert / transpose
__global__ __launch_bounds__(256) void cvt_bf16_kernel(
    const float* __restrict__ s, bf16_t* __restrict__ d, long n) {
  const long i = (long)blockIdx.x * 256 + threadIdx.x;
  if (i < n) d[i] = (bf16_t)s[i];
}

// WT[n][k] = bf16(W[k][n]) via LDS 32x32 tile
__global__ __launch_bounds__(256) void transpose_cvt_kernel(
    const float* __restrict__ W, bf16_t* __restrict__ WT, int K, int N) {
  __shared__ float tile[32][33];
  const int tx = threadIdx.x & 31, ty = threadIdx.x >> 5;
  const int n0 = blockIdx.x * 32, k0 = blockIdx.y * 32;
  for (int r = ty; r < 32; r += 8)
    tile[r][tx] = W[(long)(k0 + r) * N + n0 + tx];
  __syncthreads();
  for (int r = ty; r < 32; r += 8)
    WT[(long)(n0 + r) * K + k0 + tx] = (bf16_t)tile[tx][r];
}

// ---------------------------------------------------------------- GEMM (bf16 in, async-to-LDS, double buffered)
// C[M,N] = A[M,K] * BT[N,K]^T ; 128x128 block tile, K-step 32,
// 8 waves each own 64x32 of C (4x2 16x16 accumulators, 8 WMMA / step).
__global__ __launch_bounds__(256) void wmma_gemm_async_kernel(
    const bf16_t* __restrict__ A, const bf16_t* __restrict__ BT,
    float* __restrict__ C, int M, int N, int K) {
  __shared__ __align__(16) bf16_t Asm[2][128 * 40];
  __shared__ __align__(16) bf16_t Bsm[2][128 * 40];
  const int tid  = threadIdx.x;
  const int wave = tid >> 5, lane = tid & 31;
  const int grp  = lane >> 4, col = lane & 15;
  const int m0 = blockIdx.y * 128, n0 = blockIdx.x * 128;
  const int wm = (wave & 1) * 64;
  const int wn = (wave >> 1) * 32;
  v8f acc[4][2] = {};

  // each thread copies 2 16B chunks of A and 2 of B per buffer fill
  auto issue = [&](int bsel, int k0) {
#pragma unroll
    for (int cc = 0; cc < 2; ++cc) {
      const int c = tid * 2 + cc;
      const int row = c >> 2, part = c & 3;
      async_ld_b128(lds_addr_of(&Asm[bsel][row * 40 + part * 8]),
                    A + (long)(m0 + row) * K + k0 + part * 8);
      async_ld_b128(lds_addr_of(&Bsm[bsel][row * 40 + part * 8]),
                    BT + (long)(n0 + row) * K + k0 + part * 8);
    }
  };
  auto compute = [&](int bsel) {
    const bf16_t* Ab = Asm[bsel];
    const bf16_t* Bb = Bsm[bsel];
    v16bf afr[4], bfr[2];
#pragma unroll
    for (int i = 0; i < 4; ++i) afr[i] = ld_frag(Ab, 40, wm + 16 * i, 0);
#pragma unroll
    for (int j = 0; j < 2; ++j) bfr[j] = ld_frag(Bb, 40, wn + 16 * j, 0);
#pragma unroll
    for (int i = 0; i < 4; ++i)
#pragma unroll
      for (int j = 0; j < 2; ++j)
        acc[i][j] = wmma_bf16(afr[i], bfr[j], acc[i][j]);
  };

  int buf = 0;
  issue(0, 0);                                     // ASYNCcnt = 4 (per wave)
  int k0 = 0;
  for (; k0 + 32 < K; k0 += 32) {                  // branch-free steady state
    issue(buf ^ 1, k0 + 32);                       // ASYNCcnt -> 8
    asm volatile("s_wait_asynccnt 0x4" ::: "memory");
    __syncthreads();
    compute(buf);
    __syncthreads();
    buf ^= 1;
  }
  asm volatile("s_wait_asynccnt 0x0" ::: "memory"); // last tile
  __syncthreads();
  compute(buf);
#pragma unroll
  for (int i = 0; i < 4; ++i)
#pragma unroll
    for (int j = 0; j < 2; ++j) {
      CFrag c; c.v = acc[i][j];
#pragma unroll
      for (int r = 0; r < 8; ++r)
        C[(long)(m0 + wm + 16 * i + r + grp * 8) * N + n0 + wn + 16 * j + col] = c.f[r];
    }
}

// ---------------------------------------------------------------- beta / g gates
__global__ __launch_bounds__(256) void ba_gate_kernel(
    const float* __restrict__ X, const float* __restrict__ Wba,
    const float* __restrict__ dtb, const float* __restrict__ Alog,
    float* __restrict__ g_out, float* __restrict__ beta_out) {
  const int wid  = (int)((blockIdx.x * 256 + threadIdx.x) >> 5);
  const int lane = threadIdx.x & 31;
  const int h = wid & 15, tok = wid >> 4;
  if (tok >= NTOK) return;
  float s0 = 0.f, s1 = 0.f;
  for (int c = lane; c < HH; c += 32) {
    const float x = X[(long)tok * HH + c];
    s0 += x * Wba[c * (NH * 2) + h * 2];
    s1 += x * Wba[c * (NH * 2) + h * 2 + 1];
  }
#pragma unroll
  for (int off = 16; off > 0; off >>= 1) {
    s0 += __shfl_xor(s0, off, 32);
    s1 += __shfl_xor(s1, off, 32);
  }
  if (lane == 0) {
    const float beta = 1.f / (1.f + expf(-s0));
    const float xg = s1 + dtb[h];
    const float sp = (xg > 20.f) ? xg : log1pf(expf(xg));
    const float g  = -expf(Alog[h]) * sp;
    const int b = tok / SS, t = tok % SS;
    const long idx = (long)(b * NH + h) * SS + t;
    g_out[idx] = g;
    beta_out[idx] = beta;
  }
}

// ---------------------------------------------------------------- depthwise conv + silu
__global__ __launch_bounds__(256) void conv_silu_kernel(
    const float* __restrict__ qkvz, const float* __restrict__ cw,
    float* __restrict__ qc, float* __restrict__ kc, float* __restrict__ vc) {
  const long gid = (long)blockIdx.x * 256 + threadIdx.x;
  if (gid >= (long)NTOK * CONVC) return;
  const int c   = (int)(gid % CONVC);
  const int tok = (int)(gid / CONVC);
  const int b = tok / SS, t = tok % SS;
  const int sec = c >> 11;
  const int cc  = c & 2047;
  const int h = cc >> 7, d = cc & 127;
  const int colq = h * 512 + sec * 128 + d;
  float acc = 0.f;
#pragma unroll
  for (int kk = 0; kk < KCONV; ++kk) {
    const int tt = t - (KCONV - 1) + kk;
    const float x = (tt >= 0) ? qkvz[(long)(b * SS + tt) * QKVZN + colq] : 0.f;
    acc += cw[c * KCONV + kk] * x;
  }
  const float y = acc / (1.f + expf(-acc));
  const long dst = ((long)(b * NH + h) * SS + t) * 128 + d;
  if (sec == 0)      qc[dst] = y;
  else if (sec == 1) kc[dst] = y;
  else               vc[dst] = y;
}

// ---------------------------------------------------------------- prep
// outputs: qb16 = bf16(l2norm(q)*dk^-0.5), kb16 = bf16(l2norm(k)),
//          wb16 = bf16(-inv@(k*beta*e^gc)), u_buf = f32 inv@(v*beta), gc_buf
// dyn-LDS (bytes):
//  [0      ,16896) kb   bf16[64][132]
//  [16896  ,33792) kn   bf16[64][132] -> inv f32[64][66] -> trT bf16[128][66]
//  [33792  ,50688) Tm   f32[64][66]   -> invb bf16[64][66]
//  [50688  ,52224) scratch
#define PREP_LDS 52224
__global__ __launch_bounds__(256) void prep_kernel(
    const float* __restrict__ qc, const float* __restrict__ kc,
    const float* __restrict__ vc,
    const float* __restrict__ g_buf, const float* __restrict__ beta_buf,
    float* __restrict__ gc_buf, float* __restrict__ u_buf,
    bf16_t* __restrict__ qb16, bf16_t* __restrict__ kb16,
    bf16_t* __restrict__ wb16) {
  extern __shared__ char smem[];
  bf16_t* kb   = (bf16_t*)smem;
  bf16_t* kn   = (bf16_t*)(smem + 16896);
  float*  inv  = (float*)(smem + 16896);
  bf16_t* trT  = (bf16_t*)(smem + 16896);
  float*  Tm   = (float*)(smem + 33792);
  bf16_t* invb = (bf16_t*)(smem + 33792);
  float*  sc   = (float*)(smem + 50688);
  float* gcs = sc; float* egs = sc + 64; float* bet = sc + 128; float* rsq = sc + 192;
  const int tid = threadIdx.x, wave = tid >> 5, lane = tid & 31;
  const int grp = lane >> 4, col = lane & 15;
  const int bh = blockIdx.y, chunk = blockIdx.x;
  const long rowbase = (long)bh * SS + chunk * CHKSZ;
  const float* kg = kc + rowbase * DKD;
  const float* qg = qc + rowbase * DKD;
  const float* vg = vc + rowbase * DVD;

  if (tid < CHKSZ) {
    gcs[tid] = g_buf[rowbase + tid];
    bet[tid] = beta_buf[rowbase + tid];
  }
  __syncthreads();
  if (tid == 0) {
    float run = 0.f;
    for (int i = 0; i < CHKSZ; ++i) { run += gcs[i]; gcs[i] = run; }
  }
  __syncthreads();
  if (tid < CHKSZ) {
    egs[tid] = expf(gcs[tid]);
    gc_buf[rowbase + tid] = gcs[tid];
  }
  for (int rr = 0; rr < 8; ++rr) {
    const int r = wave * 8 + rr;
    float s = 0.f;
    for (int d = lane; d < DKD; d += 32) { const float x = kg[r * DKD + d]; s += x * x; }
#pragma unroll
    for (int off = 16; off > 0; off >>= 1) s += __shfl_xor(s, off, 32);
    if (lane == 0) rsq[r] = rsqrtf(s + 1e-6f);
  }
  __syncthreads();
  for (int i = tid; i < CHKSZ * DKD; i += 256) {
    const int r = i >> 7, d = i & 127;
    const float v = kg[i] * rsq[r];
    kb16[rowbase * DKD + i] = (bf16_t)v;
    kn[r * 132 + d] = (bf16_t)v;
    kb[r * 132 + d] = (bf16_t)(v * bet[r]);
  }
  __syncthreads();
  for (int rr = 0; rr < 8; ++rr) {
    const int r = wave * 8 + rr;
    float s = 0.f;
    for (int d = lane; d < DKD; d += 32) { const float x = qg[r * DKD + d]; s += x * x; }
#pragma unroll
    for (int off = 16; off > 0; off >>= 1) s += __shfl_xor(s, off, 32);
    if (lane == 0) rsq[r] = rsqrtf(s + 1e-6f) * 0.088388347648318447f;
  }
  __syncthreads();
  for (int i = tid; i < CHKSZ * DKD; i += 256)
    qb16[rowbase * DKD + i] = (bf16_t)(qg[i] * rsq[i >> 7]);
  {
    const int wm = (wave & 3) * 16;
    const int wnb = (wave >> 2) * 32;
    v8f a0 = {}; v8f a1 = {};
    for (int k0 = 0; k0 < DKD; k0 += 32) {
      v16bf a = ld_frag(kb, 132, wm, k0);
      a0 = wmma_bf16(a, ld_frag(kn, 132, wnb, k0), a0);
      a1 = wmma_bf16(a, ld_frag(kn, 132, wnb + 16, k0), a1);
    }
    CFrag c0, c1; c0.v = a0; c1.v = a1;
#pragma unroll
    for (int r = 0; r < 8; ++r) {
      const int i = wm + r + grp * 8;
      const int j0 = wnb + col, j1 = wnb + 16 + col;
      Tm[i * 66 + j0] = (i > j0) ? (-c0.f[r]) * expf(gcs[i] - gcs[j0]) : 0.f;
      Tm[i * 66 + j1] = (i > j1) ? (-c1.f[r]) * expf(gcs[i] - gcs[j1]) : 0.f;
    }
  }
  __syncthreads();
  for (int i = tid; i < CHKSZ * CHKSZ; i += 256) {
    const int r = i >> 6, c2 = i & 63;
    inv[r * 66 + c2] = (r == c2) ? 1.f : 0.f;
  }
  __syncthreads();
  for (int i = 1; i < CHKSZ; ++i) {
    if (tid < i) {
      const int c2 = tid;
      float s = 0.f;
      for (int j = c2; j < i; ++j) s += Tm[i * 66 + j] * inv[j * 66 + c2];
      inv[i * 66 + c2] += s;
    }
    __syncthreads();
  }
  for (int i = tid; i < CHKSZ * CHKSZ; i += 256) {
    const int r = i >> 6, c2 = i & 63;
    invb[r * 66 + c2] = (bf16_t)inv[r * 66 + c2];
  }
  __syncthreads();
  for (int i = tid; i < DKD * CHKSZ; i += 256) {
    const int d = i >> 6, j = i & 63;
    trT[d * 66 + j] = (bf16_t)((float)kb[j * 132 + d] * egs[j]);
  }
  __syncthreads();
  for (int q4 = 0; q4 < 4; ++q4) {
    const int s = wave * 4 + q4;
    const int m = (s >> 3) * 16, n = (s & 7) * 16;
    v8f acc = {};
    acc = wmma_bf16(ld_frag(invb, 66, m, 0),  ld_frag(trT, 66, n, 0),  acc);
    acc = wmma_bf16(ld_frag(invb, 66, m, 32), ld_frag(trT, 66, n, 32), acc);
    CFrag c; c.v = acc;
#pragma unroll
    for (int r = 0; r < 8; ++r)       // store NEGATED w as bf16 (scan uses -w)
      wb16[(rowbase + m + r + grp * 8) * DKD + n + col] = (bf16_t)(-c.f[r]);
  }
  __syncthreads();
  for (int i = tid; i < DVD * CHKSZ; i += 256) {
    const int d = i >> 6, j = i & 63;
    trT[d * 66 + j] = (bf16_t)(vg[(long)j * DVD + d] * bet[j]);
  }
  __syncthreads();
  for (int q4 = 0; q4 < 4; ++q4) {
    const int s = wave * 4 + q4;
    const int m = (s >> 3) * 16, n = (s & 7) * 16;
    v8f acc = {};
    acc = wmma_bf16(ld_frag(invb, 66, m, 0),  ld_frag(trT, 66, n, 0),  acc);
    acc = wmma_bf16(ld_frag(invb, 66, m, 32), ld_frag(trT, 66, n, 32), acc);
    CFrag c; c.v = acc;
#pragma unroll
    for (int r = 0; r < 8; ++r)
      u_buf[(rowbase + m + r + grp * 8) * DVD + n + col] = c.f[r];
  }
}

// ---------------------------------------------------------------- scan: chunk recurrence
// q/k/w tiles (bf16 global) stream in via double-buffered async-to-LDS.
// dyn-LDS (bytes):
//  [0      , 33792) Ss    f32 [128][66]
//  [33792  , 50688) SbT   bf16[64][132]
//  [50688  , 67584) kdT   bf16[128][66]
//  [67584  , 76032) vnT   bf16[64][66]
//  [76032  , 84480) attnb bf16[64][66]
//  [84480  , 86016) scratch
//  [86016  ,120832) qt[2]  bf16[64][136]
//  [120832 ,155648) kt[2]  bf16[64][136]
//  [155648 ,190464) wn[2]  bf16[64][136]
#define SCAN_LDS 190464
#define TSTRIDE 136
#define TSIZE   (64 * TSTRIDE)
__global__ __launch_bounds__(256) void scan_kernel(
    const bf16_t* __restrict__ qb16, const bf16_t* __restrict__ kb16,
    const float* __restrict__ gc_buf, const float* __restrict__ u_buf,
    const bf16_t* __restrict__ wb16, float* __restrict__ core) {
  extern __shared__ char smem[];
  float*  Ss    = (float*)smem;
  bf16_t* SbT   = (bf16_t*)(smem + 33792);
  bf16_t* kdT   = (bf16_t*)(smem + 50688);
  bf16_t* vnT   = (bf16_t*)(smem + 67584);
  bf16_t* attnb = (bf16_t*)(smem + 76032);
  float*  sc    = (float*)(smem + 84480);
  bf16_t* qtB   = (bf16_t*)(smem + 86016);
  bf16_t* ktB   = (bf16_t*)(smem + 120832);
  bf16_t* wnB   = (bf16_t*)(smem + 155648);
  float* gcs = sc; float* egs = sc + 64; float* kds = sc + 128;
  const int tid = threadIdx.x, wave = tid >> 5, lane = tid & 31;
  const int grp = lane >> 4, col = lane & 15;
  const int half = blockIdx.x;
  const int bh = blockIdx.y;
  const int b = bh >> 4, h = bh & 15;
  for (int i = tid; i < 128 * 66; i += 256) Ss[i] = 0.f;

  // async fill of one chunk's q/k/w tiles: 12 async ops per thread / wave
  auto issue_tiles = [&](int bsel, int cnext) {
    const long rb = ((long)bh * SS + (long)cnext * CHKSZ) * DKD;
    bf16_t* qd = qtB + bsel * TSIZE;
    bf16_t* kd = ktB + bsel * TSIZE;
    bf16_t* wd = wnB + bsel * TSIZE;
#pragma unroll
    for (int cc = 0; cc < 4; ++cc) {
      const int ch = tid * 4 + cc;          // 0..1023 : 16B chunks of a 64x128 tile
      const int r = ch >> 4, part = ch & 15;
      const long go = rb + r * DKD + part * 8;
      const int  lo = r * TSTRIDE + part * 8;
      async_ld_b128(lds_addr_of(qd + lo), qb16 + go);
      async_ld_b128(lds_addr_of(kd + lo), kb16 + go);
      async_ld_b128(lds_addr_of(wd + lo), wb16 + go);
    }
  };

  int buf = 0;
  issue_tiles(0, 0);                            // ASYNCcnt = 12
  for (int cix = 0; cix < NCHUNK; ++cix) {
    const long rowbase = (long)bh * SS + cix * CHKSZ;
    __syncthreads();                            // prev state update done
    const bool more = (cix + 1) < NCHUNK;
    if (more) {
      issue_tiles(buf ^ 1, cix + 1);            // ASYNCcnt -> 24
      __builtin_prefetch(u_buf + (rowbase + CHKSZ) * DVD + tid * 32, 0, 1);
    }
    if (tid < CHKSZ) {
      const float g = gc_buf[rowbase + tid];
      gcs[tid] = g;
      egs[tid] = expf(g);
    }
    __syncthreads();
    if (tid < CHKSZ) kds[tid] = expf(gcs[63] - gcs[tid]);
    for (int i = tid; i < 64 * 128; i += 256) {
      const int dc = i >> 7, d = i & 127;
      SbT[dc * 132 + d] = (bf16_t)Ss[d * 66 + dc];
    }
    if (more) asm volatile("s_wait_asynccnt 0xc" ::: "memory");  // cur tiles landed
    else      asm volatile("s_wait_asynccnt 0x0" ::: "memory");
    __syncthreads();
    bf16_t* qt = qtB + buf * TSIZE;
    bf16_t* kt = ktB + buf * TSIZE;
    bf16_t* wn = wnB + buf * TSIZE;
    // attnb = mask(q @ k^T) * decay
    {
      const int wm = (wave & 3) * 16;
      const int wnb = (wave >> 2) * 32;
      v8f a0 = {}; v8f a1 = {};
      for (int k0 = 0; k0 < DKD; k0 += 32) {
        v16bf a = ld_frag(qt, TSTRIDE, wm, k0);
        a0 = wmma_bf16(a, ld_frag(kt, TSTRIDE, wnb, k0), a0);
        a1 = wmma_bf16(a, ld_frag(kt, TSTRIDE, wnb + 16, k0), a1);
      }
      CFrag c0, c1; c0.v = a0; c1.v = a1;
#pragma unroll
      for (int r = 0; r < 8; ++r) {
        const int i = wm + r + grp * 8;
        const int j0 = wnb + col, j1 = wnb + 16 + col;
        attnb[i * 66 + j0] = (bf16_t)((i >= j0) ? c0.f[r] * expf(gcs[i] - gcs[j0]) : 0.f);
        attnb[i * 66 + j1] = (bf16_t)((i >= j1) ? c1.f[r] * expf(gcs[i] - gcs[j1]) : 0.f);
      }
    }
    __syncthreads();
    // v_new = u + (-w) @ S -> vnT (transposed bf16)
    for (int q2 = 0; q2 < 2; ++q2) {
      const int s = wave * 2 + q2;
      const int m = (s >> 2) * 16, n = (s & 3) * 16;
      CFrag c;
#pragma unroll
      for (int r = 0; r < 8; ++r)
        c.f[r] = u_buf[(rowbase + m + r + grp * 8) * DVD + half * 64 + n + col];
      v8f acc = c.v;
      for (int k0 = 0; k0 < DKD; k0 += 32)
        acc = wmma_bf16(ld_frag(wn, TSTRIDE, m, k0), ld_frag(SbT, 132, n, k0), acc);
      c.v = acc;
#pragma unroll
      for (int r = 0; r < 8; ++r)
        vnT[(n + col) * 66 + m + r + grp * 8] = (bf16_t)c.f[r];
    }
    __syncthreads();
    // q <- q * exp(gc), and kdT[d][r] = k[r][d] * exp(g_last - gc_r) from LDS k tile
    for (int i = tid; i < CHKSZ * DKD; i += 256) {
      const int r = i >> 7, d = i & 127;
      qt[r * TSTRIDE + d] = (bf16_t)((float)qt[r * TSTRIDE + d] * egs[r]);
      kdT[d * 66 + r] = (bf16_t)((float)kt[r * TSTRIDE + d] * kds[r]);
    }
    __syncthreads();
    // out = (q e^gc) @ S + attn @ v_new
    for (int q2 = 0; q2 < 2; ++q2) {
      const int s = wave * 2 + q2;
      const int m = (s >> 2) * 16, n = (s & 3) * 16;
      v8f acc = {};
      for (int k0 = 0; k0 < DKD; k0 += 32)
        acc = wmma_bf16(ld_frag(qt, TSTRIDE, m, k0), ld_frag(SbT, 132, n, k0), acc);
      acc = wmma_bf16(ld_frag(attnb, 66, m, 0),  ld_frag(vnT, 66, n, 0),  acc);
      acc = wmma_bf16(ld_frag(attnb, 66, m, 32), ld_frag(vnT, 66, n, 32), acc);
      CFrag c; c.v = acc;
#pragma unroll
      for (int r = 0; r < 8; ++r) {
        const int i = m + r + grp * 8;
        const long tok = (long)b * SS + cix * CHKSZ + i;
        core[tok * (NH * DVD) + h * DVD + half * 64 + n + col] = c.f[r];
      }
    }
    __syncthreads();
    // S = S * exp(g_last) + k_dec^T @ v_new
    {
      const float el = egs[63];
      for (int q4 = 0; q4 < 4; ++q4) {
        const int s = wave * 4 + q4;
        const int m = (s >> 2) * 16, n = (s & 3) * 16;
        CFrag c;
#pragma unroll
        for (int r = 0; r < 8; ++r)
          c.f[r] = Ss[(m + r + grp * 8) * 66 + n + col] * el;
        v8f acc = c.v;
        acc = wmma_bf16(ld_frag(kdT, 66, m, 0),  ld_frag(vnT, 66, n, 0),  acc);
        acc = wmma_bf16(ld_frag(kdT, 66, m, 32), ld_frag(vnT, 66, n, 32), acc);
        c.v = acc;
#pragma unroll
        for (int r = 0; r < 8; ++r)
          Ss[(m + r + grp * 8) * 66 + n + col] = c.f[r];
      }
    }
    buf ^= 1;
  }
}

// ---------------------------------------------------------------- gated RMSNorm (in place)
__global__ __launch_bounds__(256) void gated_norm_kernel(
    float* __restrict__ core, const float* __restrict__ qkvz,
    const float* __restrict__ nw) {
  const int wid  = (int)((blockIdx.x * 256 + threadIdx.x) >> 5);
  const int lane = threadIdx.x & 31;
  const int h = wid & 15, tok = wid >> 4;
  if (tok >= NTOK) return;
  float x[4];
  float ss = 0.f;
#pragma unroll
  for (int i = 0; i < 4; ++i) {
    x[i] = core[(long)tok * 2048 + h * 128 + lane + i * 32];
    ss += x[i] * x[i];
  }
#pragma unroll
  for (int off = 16; off > 0; off >>= 1) ss += __shfl_xor(ss, off, 32);
  const float rinv = rsqrtf(ss * (1.f / 128.f) + 1e-6f);
#pragma unroll
  for (int i = 0; i < 4; ++i) {
    const int d = lane + i * 32;
    const float z = qkvz[(long)tok * QKVZN + h * 512 + 384 + d];
    const float sz = z / (1.f + expf(-z));
    core[(long)tok * 2048 + h * 128 + d] = nw[d] * x[i] * rinv * sz;
  }
}

// ---------------------------------------------------------------- launch
extern "C" void kernel_launch(void* const* d_in, const int* in_sizes, int n_in,
                              void* d_out, int out_size, void* d_ws, size_t ws_size,
                              hipStream_t stream) {
  (void)in_sizes; (void)n_in; (void)out_size; (void)ws_size;
  const float* X     = (const float*)d_in[0];
  const float* Wqkvz = (const float*)d_in[1];
  const float* Wba   = (const float*)d_in[2];
  const float* convw = (const float*)d_in[3];
  const float* dtb   = (const float*)d_in[4];
  const float* Alog  = (const float*)d_in[5];
  const float* nw    = (const float*)d_in[6];
  const float* Wout  = (const float*)d_in[7];
  float* out = (float*)d_out;

  char* ws = (char*)d_ws;
  size_t off = 0;
  float* qkvz    = (float*)(ws + off);  off += (size_t)NTOK * QKVZN * 4;
  float* qc      = (float*)(ws + off);  off += (size_t)BB * NH * SS * DKD * 4;
  float* kc      = (float*)(ws + off);  off += (size_t)BB * NH * SS * DKD * 4;
  float* vc      = (float*)(ws + off);  off += (size_t)BB * NH * SS * DVD * 4;
  float* g_buf   = (float*)(ws + off);  off += (size_t)BB * NH * SS * 4;
  float* b_buf   = (float*)(ws + off);  off += (size_t)BB * NH * SS * 4;
  float* gc_buf  = (float*)(ws + off);  off += (size_t)BB * NH * SS * 4;
  float* u_buf   = (float*)(ws + off);  off += (size_t)BB * NH * SS * DVD * 4;
  float* core    = (float*)(ws + off);  off += (size_t)NTOK * NH * DVD * 4;
  bf16_t* Xbf    = (bf16_t*)(ws + off); off += (size_t)NTOK * HH * 2;
  bf16_t* WqkvzT = (bf16_t*)(ws + off); off += (size_t)QKVZN * HH * 2;
  bf16_t* corebf = (bf16_t*)(ws + off); off += (size_t)NTOK * HH * 2;
  bf16_t* WoutT  = (bf16_t*)(ws + off); off += (size_t)HH * HH * 2;
  bf16_t* qb16   = (bf16_t*)(ws + off); off += (size_t)BB * NH * SS * DKD * 2;
  bf16_t* kb16   = (bf16_t*)(ws + off); off += (size_t)BB * NH * SS * DKD * 2;
  bf16_t* wb16   = (bf16_t*)(ws + off); off += (size_t)BB * NH * SS * DKD * 2;

  // 0) bf16 conversions / weight transposes
  cvt_bf16_kernel<<<(int)(((long)NTOK * HH + 255) / 256), 256, 0, stream>>>(
      X, Xbf, (long)NTOK * HH);
  transpose_cvt_kernel<<<dim3(QKVZN / 32, HH / 32), 256, 0, stream>>>(
      Wqkvz, WqkvzT, HH, QKVZN);
  transpose_cvt_kernel<<<dim3(HH / 32, HH / 32), 256, 0, stream>>>(
      Wout, WoutT, HH, HH);
  // 1) qkvz = X @ W_qkvz (async-LDS double-buffered bf16 WMMA)
  wmma_gemm_async_kernel<<<dim3(QKVZN / 128, NTOK / 128), 256, 0, stream>>>(
      Xbf, WqkvzT, qkvz, NTOK, QKVZN, HH);
  // 2) beta, g gates
  ba_gate_kernel<<<(NTOK * NH) / 8, 256, 0, stream>>>(X, Wba, dtb, Alog, g_buf, b_buf);
  // 3) depthwise causal conv + silu
  conv_silu_kernel<<<(NTOK * CONVC) / 256, 256, 0, stream>>>(qkvz, convw, qc, kc, vc);
  // 4) per-chunk prep (emits bf16 q/k/-w tiles + f32 u)
  prep_kernel<<<dim3(NCHUNK, BB * NH), 256, PREP_LDS, stream>>>(
      qc, kc, vc, g_buf, b_buf, gc_buf, u_buf, qb16, kb16, wb16);
  // 5) chunk-recurrent scan (DV split in 2, async double-buffered tiles)
  scan_kernel<<<dim3(2, BB * NH), 256, SCAN_LDS, stream>>>(
      qb16, kb16, gc_buf, u_buf, wb16, core);
  // 6) gated RMSNorm
  gated_norm_kernel<<<(NTOK * NH) / 8, 256, 0, stream>>>(core, qkvz, nw);
  // 7) out = core @ W_out
  cvt_bf16_kernel<<<(int)(((long)NTOK * HH + 255) / 256), 256, 0, stream>>>(
      core, corebf, (long)NTOK * HH);
  wmma_gemm_async_kernel<<<dim3(HH / 128, NTOK / 128), 256, 0, stream>>>(
      corebf, WoutT, out, NTOK, HH, HH);
}
